// LSTMSpatioTemporalAttention_10986526343972
// MI455X (gfx1250) — compile-verified, matrix operand order checked by reading, not run on hardware
//
#include <hip/hip_runtime.h>

// ---------------------------------------------------------------------------
// LSTM + spatio-temporal attention, CDNA5 (gfx1250, wave32, WMMA f16->f32)
//
// B=256, T=512, F=128, H=256, 4H=1024, O=1
//
//   (0) convert weights f32->f16, fold layer_in into LSTM0 input weights
//   (1) BN + spatial attention (rowwise, sa_W cached in LDS as f16)
//   (2) 512x LSTM0 steps: async-to-LDS staged activations, wave computes all
//       4 gate tiles (A reused 4x, register-local cell update)
//   (3) 512x LSTM1 steps (same kernel, K=256+256)
//   (4) total = h1 @ ta_hW^T + ta_hb   (2x2-tiled WMMA GEMM, LDS-staged A)
//   (5) logits = relu(total @ ta_W^T + ta_b)
//   (6) softmax over T + context (wave per (b,f), shfl_xor)
//   (7) out = ctx @ out_W^T
// ---------------------------------------------------------------------------

#define BB   256
#define TT   512
#define FF   128
#define HH   256
#define G4   1024      // 4*H
#define BT   (BB*TT)   // 131072

typedef __attribute__((ext_vector_type(16))) _Float16 v16h;
typedef __attribute__((ext_vector_type(8)))  _Float16 v8h;
typedef __attribute__((ext_vector_type(8)))  float    v8f;

union Tile16 { v16h v; v8h h[2]; };

// ---- CDNA5 async global->LDS copy (ASYNCcnt-tracked) ----------------------
__device__ __forceinline__ void async_g2l_b128(unsigned lds_off, const void* g) {
  unsigned long long ga = (unsigned long long)g;
  asm volatile("global_load_async_to_lds_b128 %0, %1, off"
               :: "v"(lds_off), "v"(ga) : "memory");
}
__device__ __forceinline__ void wait_async0() {
  asm volatile("s_wait_asynccnt 0" ::: "memory");
}

// Load a 16x32 f16 operand tile in the documented CDNA5 WMMA layout:
//   lanes 0-15 : row = r0+lane,    K halves {k0..k0+7, k0+16..k0+23}
//   lanes 16-31: row = r0+lane-16, K halves {k0+8..k0+15, k0+24..k0+31}
__device__ __forceinline__ v16h load_tile16(const _Float16* __restrict__ base,
                                            size_t ld, int r0, int k0) {
  const int lane = threadIdx.x & 31;
  const int r    = r0 + (lane & 15);
  const int koff = (lane & 16) ? 8 : 0;
  const _Float16* p = base + (size_t)r * ld + (size_t)(k0 + koff);
  Tile16 t;
  t.h[0] = *(const v8h*)(p);
  t.h[1] = *(const v8h*)(p + 16);
  return t.v;
}
// Same tile gather but from LDS (emits ds_load_b128)
__device__ __forceinline__ v16h load_tile16_lds(const _Float16* base,
                                                int ld, int k0) {
  const int lane = threadIdx.x & 31;
  const int r    = lane & 15;
  const int koff = (lane & 16) ? 8 : 0;
  const _Float16* p = base + r * ld + k0 + koff;
  Tile16 t;
  t.h[0] = *(const v8h*)(p);
  t.h[1] = *(const v8h*)(p + 16);
  return t.v;
}

__device__ __forceinline__ float sigmoidf_(float x) {
  return 1.0f / (1.0f + __expf(-x));
}

// ---------------------------------------------------------------------------
// small utility kernels
// ---------------------------------------------------------------------------
__global__ void k_zero_f32(float* p, int n) {
  for (int i = blockIdx.x * blockDim.x + threadIdx.x; i < n; i += gridDim.x * blockDim.x)
    p[i] = 0.0f;
}
__global__ void k_zero_f16(_Float16* p, int n) {
  for (int i = blockIdx.x * blockDim.x + threadIdx.x; i < n; i += gridDim.x * blockDim.x)
    p[i] = (_Float16)0.0f;
}
__global__ void k_cvt_f16(_Float16* __restrict__ dst, const float* __restrict__ src, int n) {
  for (int i = blockIdx.x * blockDim.x + threadIdx.x; i < n; i += gridDim.x * blockDim.x)
    dst[i] = (_Float16)src[i];
}
__global__ void k_add_bias(float* __restrict__ dst, const float* __restrict__ a,
                           const float* __restrict__ b, int n) {
  for (int i = blockIdx.x * blockDim.x + threadIdx.x; i < n; i += gridDim.x * blockDim.x)
    dst[i] = a[i] + b[i];
}

// Wc[n,f] = sum_h Wih0[n,h]*in_W[h,f]  (f16 out);  bc[n] = bih0+bhh0+Wih0@in_b
__global__ void k_wcomb(const float* __restrict__ Wih0, const float* __restrict__ inW,
                        const float* __restrict__ inb,  const float* __restrict__ bih0,
                        const float* __restrict__ bhh0,
                        _Float16* __restrict__ Wc, float* __restrict__ bc) {
  const int n = blockIdx.x;          // 0..1023
  const int f = threadIdx.x;         // 0..127
  float acc = 0.0f;
  for (int h = 0; h < HH; ++h)
    acc += Wih0[(size_t)n * HH + h] * inW[(size_t)h * FF + f];
  Wc[(size_t)n * FF + f] = (_Float16)acc;
  if (f == 0) {
    float s = bih0[n] + bhh0[n];
    for (int h = 0; h < HH; ++h) s += Wih0[(size_t)n * HH + h] * inb[h];
    bc[n] = s;
  }
}

// ---------------------------------------------------------------------------
// (1) BatchNorm + spatial attention, rowwise.  block=128 threads, 64 rows/block
// ---------------------------------------------------------------------------
#define SAW_LD 130   // padded f16 stride: (130/2)%64 is odd -> conflict-free
__global__ void k_bn_sa(const float* __restrict__ x,
                        const float* __restrict__ gamma, const float* __restrict__ beta,
                        const float* __restrict__ mean,  const float* __restrict__ var,
                        const _Float16* __restrict__ saW, const float* __restrict__ sab,
                        _Float16* __restrict__ xs) {
  __shared__ _Float16 sw[FF * SAW_LD];
  __shared__ float xb[FF];
  __shared__ float red[FF];
  __shared__ float scale[FF];
  __shared__ float shift[FF];

  const int f = threadIdx.x;  // 128 threads
  for (int i = f; i < FF * FF; i += FF)
    sw[(i / FF) * SAW_LD + (i % FF)] = saW[i];
  const float sc = gamma[f] * __frsqrt_rn(var[f] + 1e-5f);
  scale[f] = sc;
  shift[f] = beta[f] - mean[f] * sc;
  __syncthreads();

  const int row0 = blockIdx.x * 64;
  const float sabf = sab[f];
  for (int rr = 0; rr < 64; ++rr) {
    const size_t row = (size_t)(row0 + rr);
    const float xv = x[row * FF + f] * scale[f] + shift[f];
    xb[f] = xv;
    __syncthreads();
    float acc = sabf;
    #pragma unroll 4
    for (int k = 0; k < FF; ++k)
      acc += xb[k] * (float)sw[f * SAW_LD + k];
    const float sg = sigmoidf_(acc);
    red[f] = sg; __syncthreads();
    for (int s = 64; s > 0; s >>= 1) {
      if (f < s) red[f] = fmaxf(red[f], red[f + s]);
      __syncthreads();
    }
    const float m = red[0]; __syncthreads();
    const float e = __expf(sg - m);
    red[f] = e; __syncthreads();
    for (int s = 64; s > 0; s >>= 1) {
      if (f < s) red[f] += red[f + s];
      __syncthreads();
    }
    const float inv = 1.0f / red[0]; __syncthreads();
    xs[row * FF + f] = (_Float16)(xv * e * inv);
  }
}

// ---------------------------------------------------------------------------
// (2)/(3) LSTM step kernel. grid = (B/16, H/64), block = 128 (4 waves).
// Block stages X_t rows (16 x Ka) and h_{t-1} rows (16 x H) into LDS with
// async-to-LDS; wave w owns h-tile w and computes ALL FOUR gate tiles,
// reusing the A fetch 4x. Gate values land in matching lanes/acc slots, so
// the cell update is register-local (no LDS exchange).
// ---------------------------------------------------------------------------
__global__ void k_lstm_step(const _Float16* __restrict__ X, size_t xld, int Ka,
                            const _Float16* __restrict__ Wa,   // [4H, Ka] f16
                            const _Float16* __restrict__ hm1,  // [B, H]   f16
                            const _Float16* __restrict__ Whh,  // [4H, H]  f16
                            const float* __restrict__ bias,    // [4H]
                            float* __restrict__ c,             // [B, H]
                            _Float16* __restrict__ hout) {     // [B, H]
  extern __shared__ _Float16 smem[];
  const int SX = Ka + 8;                 // padded LDS strides (16B aligned)
  const int SH = HH + 8;
  _Float16* sX = smem;                   // 16 x SX
  _Float16* sH = smem + 16 * SX;         // 16 x SH

  const int tid  = threadIdx.x;
  const int wave = tid >> 5;
  const int lane = tid & 31;
  const int b0 = blockIdx.x * 16;
  const int h0 = blockIdx.y * 64 + wave * 16;

  // ---- async stage A operands into LDS (ASYNCcnt path) ----
  {
    const int nchX = 16 * (Ka >> 3);     // 8-half (16B) chunks
    for (int ch = tid; ch < nchX; ch += 128) {
      const int row = ch / (Ka >> 3);
      const int kc  = (ch % (Ka >> 3)) << 3;
      async_g2l_b128((unsigned)((row * SX + kc) * 2 + 0),
                     X + (size_t)(b0 + row) * xld + kc);
    }
    const int base = 16 * SX * 2;
    for (int ch = tid; ch < 16 * (HH >> 3); ch += 128) {
      const int row = ch >> 5;           // HH/8 = 32 chunks per row
      const int kc  = (ch & 31) << 3;
      async_g2l_b128((unsigned)(base + (row * SH + kc) * 2),
                     hm1 + (size_t)(b0 + row) * HH + kc);
    }
    wait_async0();
  }
  __syncthreads();

  v8f acc[4] = {};
  // input contribution, K = Ka
  for (int k0 = 0; k0 < Ka; k0 += 32) {
    v16h a = load_tile16_lds(sX, SX, k0);
    #pragma unroll
    for (int g = 0; g < 4; ++g) {
      __builtin_prefetch(Wa + (size_t)(g * HH + h0) * Ka + k0 + 64, 0, 1);
      v16h w = load_tile16(Wa, (size_t)Ka, g * HH + h0, k0);
      acc[g] = __builtin_amdgcn_wmma_f32_16x16x32_f16(false, a, false, w,
                                                      (short)0, acc[g], false, false);
    }
  }
  // recurrent contribution, K = H
  for (int k0 = 0; k0 < HH; k0 += 32) {
    v16h a = load_tile16_lds(sH, SH, k0);
    #pragma unroll
    for (int g = 0; g < 4; ++g) {
      __builtin_prefetch(Whh + (size_t)(g * HH + h0) * HH + k0 + 64, 0, 1);
      v16h w = load_tile16(Whh, (size_t)HH, g * HH + h0, k0);
      acc[g] = __builtin_amdgcn_wmma_f32_16x16x32_f16(false, a, false, w,
                                                      (short)0, acc[g], false, false);
    }
  }

  // register-local cell update: lane (r, col) holds i/f/g/o of same element
  const int col = lane & 15;
  const int rbase = (lane & 16) ? 8 : 0;
  const int h = h0 + col;
  const float bi = bias[0 * HH + h];
  const float bf = bias[1 * HH + h];
  const float bg = bias[2 * HH + h];
  const float bo = bias[3 * HH + h];
  #pragma unroll
  for (int r = 0; r < 8; ++r) {
    const int b = b0 + rbase + r;
    const size_t idx = (size_t)b * HH + h;
    const float cp = c[idx];
    const float cn = sigmoidf_(acc[1][r] + bf) * cp +
                     sigmoidf_(acc[0][r] + bi) * tanhf(acc[2][r] + bg);
    const float hn = sigmoidf_(acc[3][r] + bo) * tanhf(cn);
    c[idx] = cn;
    hout[idx] = (_Float16)hn;
  }
}

// ---------------------------------------------------------------------------
// (4)/(5) WMMA GEMM: C[M,128] = act(A[M,K] @ W[128,K]^T + bias)
// grid = M/32, block = 128 (4 waves). Block stages A (32 rows x K) in LDS via
// async-to-LDS; wave w computes a 2m x 2n register tile at n0 = w*32.
// ---------------------------------------------------------------------------
__global__ void k_gemm_n128(const _Float16* __restrict__ A, size_t lda,
                            const _Float16* __restrict__ W, int K,
                            const float* __restrict__ bias,
                            float* __restrict__ Cf, _Float16* __restrict__ Ch,
                            int relu) {
  extern __shared__ _Float16 smem[];
  const int SA = K + 8;
  const int tid  = threadIdx.x;
  const int wave = tid >> 5;
  const int lane = tid & 31;
  const int m0 = blockIdx.x * 32;
  const int n0 = wave * 32;

  // stage 32 rows of A
  const int cpr = K >> 3;                 // 16B chunks per row
  for (int ch = tid; ch < 32 * cpr; ch += 128) {
    const int row = ch / cpr;
    const int kc  = (ch % cpr) << 3;
    async_g2l_b128((unsigned)((row * SA + kc) * 2),
                   A + (size_t)(m0 + row) * lda + kc);
  }
  wait_async0();
  __syncthreads();

  v8f acc[4] = {};                        // [mi*2+ni]
  for (int k0 = 0; k0 < K; k0 += 32) {
    v16h a0 = load_tile16_lds(smem,           SA, k0);
    v16h a1 = load_tile16_lds(smem + 16 * SA, SA, k0);
    __builtin_prefetch(W + (size_t)n0 * K + k0 + 64, 0, 1);
    v16h w0 = load_tile16(W, (size_t)K, n0,      k0);
    v16h w1 = load_tile16(W, (size_t)K, n0 + 16, k0);
    acc[0] = __builtin_amdgcn_wmma_f32_16x16x32_f16(false, a0, false, w0, (short)0, acc[0], false, false);
    acc[1] = __builtin_amdgcn_wmma_f32_16x16x32_f16(false, a0, false, w1, (short)0, acc[1], false, false);
    acc[2] = __builtin_amdgcn_wmma_f32_16x16x32_f16(false, a1, false, w0, (short)0, acc[2], false, false);
    acc[3] = __builtin_amdgcn_wmma_f32_16x16x32_f16(false, a1, false, w1, (short)0, acc[3], false, false);
  }

  const int col = lane & 15;
  const int rb  = (lane & 16) ? 8 : 0;
  #pragma unroll
  for (int mi = 0; mi < 2; ++mi) {
    #pragma unroll
    for (int ni = 0; ni < 2; ++ni) {
      const int n = n0 + ni * 16 + col;
      const float bv = bias[n];
      v8f av = acc[mi * 2 + ni];
      #pragma unroll
      for (int r = 0; r < 8; ++r) {
        float v = av[r] + bv;
        if (relu) v = fmaxf(v, 0.0f);
        const size_t idx = (size_t)(m0 + mi * 16 + rb + r) * FF + n;
        Cf[idx] = v;
        if (Ch) Ch[idx] = (_Float16)v;
      }
    }
  }
}

// ---------------------------------------------------------------------------
// (6) softmax over T (axis=1) + context.  one wave per (b,f) pair.
// logits/total layout: [t][b][f]  (row = t*B+b from the GEMM)
// ---------------------------------------------------------------------------
__global__ void k_softmax_ctx(const float* __restrict__ logits,
                              const float* __restrict__ total,
                              float* __restrict__ ctx) {
  const int wid  = threadIdx.x >> 5;
  const int lane = threadIdx.x & 31;
  const int pair = blockIdx.x * 8 + wid;       // 32768 pairs
  const int b = pair >> 7, f = pair & 127;

  float v[16], w[16];
  float m = -3.4e38f;
  #pragma unroll
  for (int i = 0; i < 16; ++i) {
    const int t = i * 32 + lane;
    const size_t idx = (size_t)t * (BB * FF) + (size_t)b * FF + f;
    v[i] = logits[idx];
    w[i] = total[idx];
    m = fmaxf(m, v[i]);
  }
  for (int off = 16; off > 0; off >>= 1) m = fmaxf(m, __shfl_xor(m, off));
  float s = 0.0f, cs = 0.0f;
  #pragma unroll
  for (int i = 0; i < 16; ++i) {
    const float e = __expf(v[i] - m);
    s += e; cs += e * w[i];
  }
  for (int off = 16; off > 0; off >>= 1) {
    s  += __shfl_xor(s, off);
    cs += __shfl_xor(cs, off);
  }
  if (lane == 0) ctx[(size_t)b * FF + f] = cs / s;
}

// (7) out[b] = ctx[b,:] . out_W[0,:]
__global__ void k_out(const float* __restrict__ ctx, const float* __restrict__ oW,
                      float* __restrict__ out) {
  const int b = threadIdx.x;
  float s = 0.0f;
  for (int f = 0; f < FF; ++f) s += ctx[(size_t)b * FF + f] * oW[f];
  out[b] = s;
}

// ---------------------------------------------------------------------------
extern "C" void kernel_launch(void* const* d_in, const int* in_sizes, int n_in,
                              void* d_out, int out_size, void* d_ws, size_t ws_size,
                              hipStream_t stream) {
  (void)in_sizes; (void)n_in; (void)out_size; (void)ws_size;
  const float* x     = (const float*)d_in[0];
  const float* bn_g  = (const float*)d_in[1];
  const float* bn_b  = (const float*)d_in[2];
  const float* bn_m  = (const float*)d_in[3];
  const float* bn_v  = (const float*)d_in[4];
  const float* saW   = (const float*)d_in[5];
  const float* sab   = (const float*)d_in[6];
  const float* inW   = (const float*)d_in[7];
  const float* inb   = (const float*)d_in[8];
  const float* Wih0  = (const float*)d_in[9];
  const float* Whh0  = (const float*)d_in[10];
  const float* bih0  = (const float*)d_in[11];
  const float* bhh0  = (const float*)d_in[12];
  const float* Wih1  = (const float*)d_in[13];
  const float* Whh1  = (const float*)d_in[14];
  const float* bih1  = (const float*)d_in[15];
  const float* bhh1  = (const float*)d_in[16];
  const float* tahW  = (const float*)d_in[17];
  const float* tahb  = (const float*)d_in[18];
  const float* taW   = (const float*)d_in[19];
  const float* tab   = (const float*)d_in[20];
  const float* outW  = (const float*)d_in[21];
  float* out = (float*)d_out;

  char* ws = (char*)d_ws;
  size_t off = 0;
  auto alloc = [&](size_t bytes) -> void* {
    void* p = (void*)(ws + off);
    off += (bytes + 255) & ~(size_t)255;
    return p;
  };
  _Float16* xs     = (_Float16*)alloc((size_t)BT * FF * 2);
  _Float16* h0     = (_Float16*)alloc((size_t)BT * HH * 2);
  _Float16* h1     = (_Float16*)alloc((size_t)BT * HH * 2);
  float*    total  = (float*)   alloc((size_t)BT * FF * 4);
  _Float16* totalh = (_Float16*)alloc((size_t)BT * FF * 2);
  float*    logits = (float*)   alloc((size_t)BT * FF * 4);
  float*    ctx    = (float*)   alloc((size_t)BB * FF * 4);
  _Float16* Wc     = (_Float16*)alloc((size_t)G4 * FF * 2);
  float*    bc     = (float*)   alloc((size_t)G4 * 4);
  _Float16* Whh0h  = (_Float16*)alloc((size_t)G4 * HH * 2);
  _Float16* Wih1h  = (_Float16*)alloc((size_t)G4 * HH * 2);
  _Float16* Whh1h  = (_Float16*)alloc((size_t)G4 * HH * 2);
  float*    b1     = (float*)   alloc((size_t)G4 * 4);
  _Float16* tahWh  = (_Float16*)alloc((size_t)FF * HH * 2);
  _Float16* taWh   = (_Float16*)alloc((size_t)FF * FF * 2);
  _Float16* saWh   = (_Float16*)alloc((size_t)FF * FF * 2);
  _Float16* hzero  = (_Float16*)alloc((size_t)BB * HH * 2);
  float*    c0     = (float*)   alloc((size_t)BB * HH * 4);
  float*    c1     = (float*)   alloc((size_t)BB * HH * 4);

  // (0) init + weight prep
  k_zero_f16<<<256, 256, 0, stream>>>(hzero, BB * HH);
  k_zero_f32<<<256, 256, 0, stream>>>(c0, BB * HH);
  k_zero_f32<<<256, 256, 0, stream>>>(c1, BB * HH);
  k_cvt_f16<<<512, 256, 0, stream>>>(Whh0h, Whh0, G4 * HH);
  k_cvt_f16<<<512, 256, 0, stream>>>(Wih1h, Wih1, G4 * HH);
  k_cvt_f16<<<512, 256, 0, stream>>>(Whh1h, Whh1, G4 * HH);
  k_cvt_f16<<<128, 256, 0, stream>>>(tahWh, tahW, FF * HH);
  k_cvt_f16<<<64, 256, 0, stream>>>(taWh, taW, FF * FF);
  k_cvt_f16<<<64, 256, 0, stream>>>(saWh, saW, FF * FF);
  k_add_bias<<<4, 256, 0, stream>>>(b1, bih1, bhh1, G4);
  k_wcomb<<<G4, FF, 0, stream>>>(Wih0, inW, inb, bih0, bhh0, Wc, bc);

  // (1) BN + spatial attention -> xs (f16, [b][t][f])
  k_bn_sa<<<BT / 64, FF, 0, stream>>>(x, bn_g, bn_b, bn_m, bn_v, saWh, sab, xs);

  const dim3 sgrid(BB / 16, HH / 64);
  const size_t smem0 = (size_t)(16 * (FF + 8) + 16 * (HH + 8)) * 2;
  const size_t smem1 = (size_t)(16 * (HH + 8) + 16 * (HH + 8)) * 2;
  // (2) LSTM layer 0: input X_t = xs[:, t, :] (ld = T*F), Ka = F
  for (int t = 0; t < TT; ++t) {
    const _Float16* hm1 = t ? (h0 + (size_t)(t - 1) * BB * HH) : hzero;
    k_lstm_step<<<sgrid, 128, smem0, stream>>>(xs + (size_t)t * FF, (size_t)TT * FF, FF,
                                               Wc, hm1, Whh0h, bc, c0,
                                               h0 + (size_t)t * BB * HH);
  }
  // (3) LSTM layer 1: input = h0[t] (ld = H), Ka = H
  for (int t = 0; t < TT; ++t) {
    const _Float16* hm1 = t ? (h1 + (size_t)(t - 1) * BB * HH) : hzero;
    k_lstm_step<<<sgrid, 128, smem1, stream>>>(h0 + (size_t)t * BB * HH, (size_t)HH, HH,
                                               Wih1h, hm1, Whh1h, b1, c1,
                                               h1 + (size_t)t * BB * HH);
  }

  // (4) total = h1 @ ta_hW^T + ta_hb   [t*B+b, F]
  k_gemm_n128<<<BT / 32, 128, (size_t)(32 * (HH + 8)) * 2, stream>>>(
      h1, (size_t)HH, tahWh, HH, tahb, total, totalh, 0);
  // (5) logits = relu(total @ ta_W^T + ta_b)
  k_gemm_n128<<<BT / 32, 128, (size_t)(32 * (FF + 8)) * 2, stream>>>(
      totalh, (size_t)FF, taWh, FF, tab, logits, (_Float16*)nullptr, 1);
  // (6) softmax over T + context
  k_softmax_ctx<<<(BB * FF) / 8, 256, 0, stream>>>(logits, total, ctx);
  // (7) output
  k_out<<<1, BB, 0, stream>>>(ctx, outW, out);
}
